// EnsembleFeatureLoss_OT_10737418240165
// MI455X (gfx1250) — compile-verified, matrix operand order checked by reading, not run on hardware
//
#include <hip/hip_runtime.h>

typedef __attribute__((ext_vector_type(16))) _Float16 v16h;
typedef __attribute__((ext_vector_type(8)))  float    v8f;

#define NPART 1024
#define E_ 3
#define B_ 4096
#define D_ 1024
#define M_ 5
#define N_ 256

// ---------------------------------------------------------------------------
// Kernel 1: bandwidth-bound partial dot reduction of feature*gt.
// 100 MB of reads -> float4 (global_load_b128) grid-stride, deterministic
// per-block LDS tree, partial per block into ws[0..NPART).
// ---------------------------------------------------------------------------
__global__ __launch_bounds__(256) void dot_partial_kernel(
    const float4* __restrict__ f, const float4* __restrict__ g,
    float* __restrict__ part, int n4) {
  const int tid  = threadIdx.x;
  const int gtid = blockIdx.x * blockDim.x + tid;
  const int step = gridDim.x * blockDim.x;
  float acc = 0.0f;
  for (int i = gtid; i < n4; i += step) {
    float4 a = f[i];
    float4 b = g[i];
    acc = fmaf(a.x, b.x, acc);
    acc = fmaf(a.y, b.y, acc);
    acc = fmaf(a.z, b.z, acc);
    acc = fmaf(a.w, b.w, acc);
  }
  __shared__ float red[256];
  red[tid] = acc;
  __syncthreads();
  for (int s = 128; s > 0; s >>= 1) {
    if (tid < s) red[tid] += red[tid + s];
    __syncthreads();
  }
  if (tid == 0) part[blockIdx.x] = red[0];
}

// ---------------------------------------------------------------------------
// Kernel 2: one workgroup per extractor e.
//   phase 1: row inverse-norms (wave-parallel, shfl_xor wave32 reduce)
//   phase 2: sim[5,256] = s_norm @ t_norm^T via v_wmma_f32_16x16x32_f16
//            (M padded 5->16; fragments packed per ISA 7.12.2 layout)
//   phase 3: K = exp((sim-1)/0.1), Sinkhorn on wave 0, OT value to ws.
// ---------------------------------------------------------------------------
__global__ __launch_bounds__(256) void ot_sinkhorn_kernel(
    const float* __restrict__ gtl, const float* __restrict__ fl,
    float* __restrict__ otout) {
  __shared__ float inv_t[N_];
  __shared__ float inv_s[8];
  __shared__ float sim[M_ * N_];
  __shared__ float Kl[M_ * N_];

  const int e    = blockIdx.x;
  const int tid  = threadIdx.x;
  const int lane = tid & 31;
  const int wave = tid >> 5;
  const float* gs = gtl + (size_t)e * M_ * D_;
  const float* tf = fl  + (size_t)e * N_ * D_;

  // ---- phase 1: inverse L2 norms -------------------------------------
  for (int r = wave * 32; r < wave * 32 + 32; ++r) {   // 32 rows per wave
    float ss = 0.0f;
    for (int i = lane; i < D_; i += 32) { float v = tf[r * D_ + i]; ss = fmaf(v, v, ss); }
    for (int off = 16; off > 0; off >>= 1) ss += __shfl_xor(ss, off, 32);
    if (lane == 0) inv_t[r] = 1.0f / fmaxf(sqrtf(ss), 1e-12f);
  }
  if (wave == 0) {
    for (int r = 0; r < M_; ++r) {
      float ss = 0.0f;
      for (int i = lane; i < D_; i += 32) { float v = gs[r * D_ + i]; ss = fmaf(v, v, ss); }
      for (int off = 16; off > 0; off >>= 1) ss += __shfl_xor(ss, off, 32);
      if (lane == 0) inv_s[r] = 1.0f / fmaxf(sqrtf(ss), 1e-12f);
    }
  }
  __syncthreads();

  // ---- phase 2: WMMA similarity --------------------------------------
  // 16-bit A/B fragment layout (ISA 7.12.2): lanes 0-15 carry K {0..7,16..23},
  // lanes 16-31 carry K {8..15,24..31}; VGPR j holds the K pair
  // base + (j>=4)*16 + 2*(j&3).
  const int half8  = (lane >= 16) ? 8 : 0;
  const int ma     = lane & 15;                 // padded M index (rows >=5 are zero)
  const int mclamp = (ma < M_) ? ma : 0;
  const float sm   = (ma < M_) ? inv_s[mclamp] : 0.0f;

  for (int t = wave; t < 16; t += 8) {          // 16 N-tiles over 8 waves
    const int nb   = t * 16 + ma;               // B row = column of sim
    const float tn = inv_t[nb];
    v8f c = {};
    for (int k0 = 0; k0 < D_; k0 += 32) {
      v16h a, b;
#pragma unroll
      for (int j = 0; j < 8; ++j) {
        const int k = k0 + ((j >> 2) << 4) + half8 + ((j & 3) << 1);
        a[2 * j]     = (_Float16)(gs[mclamp * D_ + k]     * sm);
        a[2 * j + 1] = (_Float16)(gs[mclamp * D_ + k + 1] * sm);
        b[2 * j]     = (_Float16)(tf[nb * D_ + k]         * tn);
        b[2 * j + 1] = (_Float16)(tf[nb * D_ + k + 1]     * tn);
      }
      c = __builtin_amdgcn_wmma_f32_16x16x32_f16(false, a, false, b,
                                                 (short)0, c, false, false);
    }
    // C/D layout: VGPR r, lanes 0-15 -> (M=r, N=lane); rows 0..4 are valid.
    if (lane < 16) {
#pragma unroll
      for (int r = 0; r < M_; ++r) sim[r * N_ + t * 16 + lane] = c[r];
    }
  }
  __syncthreads();

  // ---- phase 3: K matrix + Sinkhorn ----------------------------------
  for (int i = tid; i < M_ * N_; i += 256) Kl[i] = expf((sim[i] - 1.0f) * 10.0f);
  __syncthreads();

  if (wave == 0) {
    float cv[8];                                 // c[n], n = lane + 32*i
#pragma unroll
    for (int i = 0; i < 8; ++i) cv[i] = 1.0f;
    float rold[M_];
#pragma unroll
    for (int m = 0; m < M_; ++m) rold[m] = 1.0f;

    float err = 1e9f;
    int it = 0;
    while (it < 100 && err >= 0.01f) {
      float rnew[M_];
      float es = 0.0f;
#pragma unroll
      for (int m = 0; m < M_; ++m) {
        float acc = 0.0f;
#pragma unroll
        for (int i = 0; i < 8; ++i) acc = fmaf(Kl[m * N_ + lane + 32 * i], cv[i], acc);
        for (int off = 16; off > 0; off >>= 1) acc += __shfl_xor(acc, off, 32);
        rnew[m] = (1.0f / (float)M_) / acc;      // r = u / (K @ c)
        es += fabsf(rnew[m] - rold[m]);
        rold[m] = rnew[m];
      }
#pragma unroll
      for (int i = 0; i < 8; ++i) {              // c = v / (K^T @ r)
        const int n = lane + 32 * i;
        float s = 0.0f;
#pragma unroll
        for (int m = 0; m < M_; ++m) s = fmaf(Kl[m * N_ + n], rnew[m], s);
        cv[i] = (1.0f / (float)N_) / s;
      }
      err = es / (float)M_;                      // broadcast-identical on all lanes
      ++it;
    }

    // loss_e = sum_{m,n} r[m]*c[n]*K[m,n]*sim[m,n]
    float acc = 0.0f;
#pragma unroll
    for (int i = 0; i < 8; ++i) {
      const int n = lane + 32 * i;
#pragma unroll
      for (int m = 0; m < M_; ++m)
        acc = fmaf(rold[m] * cv[i], Kl[m * N_ + n] * sim[m * N_ + n], acc);
    }
    for (int off = 16; off > 0; off >>= 1) acc += __shfl_xor(acc, off, 32);
    if (lane == 0) otout[e] = 2.0f * acc;        // the *2 from the reference
  }
}

// ---------------------------------------------------------------------------
// Kernel 3: deterministic final combine -> scalar loss.
// ---------------------------------------------------------------------------
__global__ __launch_bounds__(256) void final_combine_kernel(
    const float* __restrict__ ws, float* __restrict__ out) {
  __shared__ float red[256];
  const int tid = threadIdx.x;
  float a = 0.0f;
  for (int i = tid; i < NPART; i += 256) a += ws[i];
  red[tid] = a;
  __syncthreads();
  for (int s = 128; s > 0; s >>= 1) {
    if (tid < s) red[tid] += red[tid + s];
    __syncthreads();
  }
  if (tid == 0) {
    const float loss_global = red[0] / (float)(E_ * B_);
    const float loc = ws[NPART] + ws[NPART + 1] + ws[NPART + 2]; // each = 2*OT_e
    out[0] = loss_global + 0.1f * (loc / (float)E_);
  }
}

extern "C" void kernel_launch(void* const* d_in, const int* in_sizes, int n_in,
                              void* d_out, int out_size, void* d_ws, size_t ws_size,
                              hipStream_t stream) {
  const float* feature = (const float*)d_in[0];   // [E,B,D] f32
  const float* gt      = (const float*)d_in[1];   // [E,B,D] f32
  const float* gtl     = (const float*)d_in[2];   // [E,M,D] f32
  const float* fl      = (const float*)d_in[3];   // [E,N,D] f32
  float* out = (float*)d_out;
  float* ws  = (float*)d_ws;

  const int n4 = in_sizes[0] / 4;                 // E*B*D / 4 float4 elements

  dot_partial_kernel<<<NPART, 256, 0, stream>>>(
      (const float4*)feature, (const float4*)gt, ws, n4);
  ot_sinkhorn_kernel<<<E_, 256, 0, stream>>>(gtl, fl, ws + NPART);
  final_combine_kernel<<<1, 256, 0, stream>>>(ws, out);
}
